// SparseMoE_10952166604902
// MI455X (gfx1250) — matrix operand order repaired and checked by reading, not compiled, not run on hardware
//
#include <hip/hip_runtime.h>
#include <hip/hip_bf16.h>

// Problem constants (from reference)
#define B_TOKENS 65536
#define D_MODEL  256
#define N_EXP    16
#define RB       (B_TOKENS / 256)   // 256 token-blocks of 256 tokens

typedef __attribute__((ext_vector_type(16))) __bf16 v16bf;
typedef __attribute__((ext_vector_type(8)))  float  v8f;
typedef __attribute__((ext_vector_type(8)))  unsigned short us8;

union FragAB { v16bf v; us8 h[2]; };

__device__ __forceinline__ unsigned short f32_bf16(float f) {
  union { float f; unsigned int u; } c; c.f = f;
  unsigned int r = c.u + 0x7fffu + ((c.u >> 16) & 1u);   // round-to-nearest-even
  return (unsigned short)(r >> 16);
}

// ---------------------------------------------------------------------------
// Kernel 1: convert weight (d, E*d) f32 -> wbT[e][n][k] bf16 (TRANSPOSED so
// the GEMM's B staging is pure b128 copies); zero block mask.
// ---------------------------------------------------------------------------
__global__ void moe_prep_kernel(const float* __restrict__ weight,
                                unsigned short* __restrict__ wbT,
                                unsigned int* __restrict__ blockmask) {
  int idx = blockIdx.x * 256 + threadIdx.x;        // 0 .. 16*256*256-1
  int e = idx >> 16;
  int n = (idx >> 8) & 255;
  int k = idx & 255;
  wbT[idx] = f32_bf16(weight[k * (N_EXP * D_MODEL) + e * D_MODEL + n]);
  if (idx < RB) blockmask[idx] = 0u;
}

// ---------------------------------------------------------------------------
// Kernel 2: gating. One workgroup handles 16 tokens; thread (tok,e) computes
// one logit. Writes probs (B x 16), bf16 x, and ORs argmax into block mask.
// ---------------------------------------------------------------------------
__global__ __launch_bounds__(256) void moe_gate_kernel(
    const float* __restrict__ x, const float* __restrict__ gate_w,
    const float* __restrict__ gate_b, float* __restrict__ probs,
    unsigned short* __restrict__ xb, unsigned int* __restrict__ blockmask) {
  __shared__ float sX[16 * D_MODEL];     // 16 KB
  __shared__ float sG[N_EXP * D_MODEL];  // 16 KB
  __shared__ float sL[16][N_EXP + 1];
  __shared__ float sGb[N_EXP];

  int tid = threadIdx.x;
  int tokBase = blockIdx.x * 16;

  for (int i = tid; i < 16 * D_MODEL; i += 256) {
    sX[i] = x[(size_t)tokBase * D_MODEL + i];
    sG[i] = gate_w[i];
  }
  if (tid < N_EXP) sGb[tid] = gate_b[tid];
  __syncthreads();

  // bf16 conversion of x (coalesced), reuses staged tile
  for (int i = tid; i < 16 * D_MODEL; i += 256)
    xb[(size_t)tokBase * D_MODEL + i] = f32_bf16(sX[i]);

  int tok = tid >> 4;
  int e   = tid & 15;
  const float* xr = &sX[tok * D_MODEL];
  const float* gr = &sG[e * D_MODEL];
  float dot = sGb[e];
  #pragma unroll 8
  for (int k = 0; k < D_MODEL; k++) dot += xr[k] * gr[k];
  sL[tok][e] = dot;
  __syncthreads();

  // softmax + argmax over 16 experts (redundant per (tok,e) thread; cheap)
  float mx = sL[tok][0];
  int am = 0;
  #pragma unroll
  for (int j = 1; j < N_EXP; j++) {
    float v = sL[tok][j];
    if (v > mx) { mx = v; am = j; }
  }
  float sum = 0.f;
  #pragma unroll
  for (int j = 0; j < N_EXP; j++) sum += __expf(sL[tok][j] - mx);
  float p = __expf(sL[tok][e] - mx) / sum;
  probs[(size_t)(tokBase + tok) * N_EXP + e] = p;
  if (e == am)
    atomicOr(&blockmask[(tokBase + tok) >> 8], 1u << am);
}

// ---------------------------------------------------------------------------
// Kernel 3: main GEMM. grid = (D/64 col-slices, RB token-blocks), 256 threads.
// Wave w owns rows [w*32, w*32+32) x 64 cols -> 2x4 WMMA C tiles (f32).
// kc-outer / expert-inner: each 256x64 A chunk is staged ONCE per kc via the
// CDNA5 async global->LDS path (GLOBAL_LOAD_ASYNC_TO_LDS_B128, ASYNCcnt) and
// reused by all 16 experts. B slices stream through a small LDS buffer with
// register prefetch of the next expert's slice overlapping the WMMA burst.
// Masked-out experts are handled by scaling with p*maskbit, keeping EXEC
// all-ones for WMMA.
// ---------------------------------------------------------------------------
__global__ __launch_bounds__(256) void moe_gemm_kernel(
    const unsigned short* __restrict__ xb, const unsigned short* __restrict__ wbT,
    const float* __restrict__ probs, const unsigned int* __restrict__ blockmask,
    float* __restrict__ out) {
  __shared__ unsigned short sA[256][72];   // 256 rows x 64 K (+8 pad) bf16, 36 KB
  __shared__ unsigned short sB[64][72];    // 64 cols x 64 K (+8 pad) bf16,  9 KB
  __shared__ float sP[256][N_EXP];         // probs for this token block,    16 KB

  const int tid      = threadIdx.x;
  const int colBase  = blockIdx.x * 64;
  const int blockRow = blockIdx.y;          // token-block index
  const unsigned int mask = blockmask[blockRow];

  for (int i = tid; i < 256 * N_EXP; i += 256)
    sP[i >> 4][i & 15] = probs[(size_t)blockRow * 256 * N_EXP + i];

  const int wave   = tid >> 5;
  const int lane   = tid & 31;
  const int laneM  = lane & 15;
  const int laneHi = lane >> 4;
  const int rowBaseW = wave * 32;

  // B staging geometry: 64 n-rows x 64 k = 512 16B chunks, 2 per thread.
  const int bN0 = (tid + 0)   >> 3;           // n row of chunk 0
  const int bK0 = ((tid + 0)  & 7) * 8;       // k col of chunk 0
  const int bN1 = (tid + 256) >> 3;
  const int bK1 = ((tid + 256) & 7) * 8;
  const unsigned short* wbTcol = wbT + (size_t)colBase * D_MODEL;

  v8f acc[2][4];
  #pragma unroll
  for (int m = 0; m < 2; m++)
    #pragma unroll
    for (int n = 0; n < 4; n++) acc[m][n] = (v8f)0.f;

  #pragma unroll 1
  for (int kc = 0; kc < 4; kc++) {
    // first expert's B slice prefetched into registers (overlaps barrier wait)
    us8 rB0 = *(const us8*)&wbTcol[(size_t)bN0 * D_MODEL + kc * 64 + bK0];
    us8 rB1 = *(const us8*)&wbTcol[(size_t)bN1 * D_MODEL + kc * 64 + bK1];

    __syncthreads();   // all waves done with previous sA/sB (and sP staged, kc==0)

    // ---- async stage A chunk (once per kc): global -> LDS, no VGPR staging.
    // Per-lane 16B transfers; tracked by ASYNCcnt.
    #pragma unroll
    for (int i = 0; i < 8; i++) {
      int c = tid + 256 * i;
      unsigned lofs = (unsigned)(size_t)&sA[c >> 3][(c & 7) * 8];
      const unsigned short* g =
          &xb[((size_t)blockRow * 256 + (c >> 3)) * D_MODEL + kc * 64 + (c & 7) * 8];
      asm volatile("global_load_async_to_lds_b128 %0, %1, off"
                   :: "v"(lofs), "v"(g)
                   : "memory");
    }

    *(us8*)&sB[bN0][bK0] = rB0;
    *(us8*)&sB[bN1][bK1] = rB1;

    asm volatile("s_wait_asynccnt 0x0" ::: "memory");   // A chunk landed in LDS
    __syncthreads();                                    // publish sA + sB

    #pragma unroll 1
    for (int e = 0; e < N_EXP; e++) {
      // prefetch next expert's B slice (issued before the WMMA burst)
      us8 nB0, nB1;
      if (e < N_EXP - 1) {
        const unsigned short* src = wbTcol + (size_t)(e + 1) * D_MODEL * D_MODEL;
        nB0 = *(const us8*)&src[(size_t)bN0 * D_MODEL + kc * 64 + bK0];
        nB1 = *(const us8*)&src[(size_t)bN1 * D_MODEL + kc * 64 + bK1];
      }

      v8f y[2][4];
      #pragma unroll
      for (int m = 0; m < 2; m++)
        #pragma unroll
        for (int n = 0; n < 4; n++) y[m][n] = (v8f)0.f;

      #pragma unroll
      for (int ks = 0; ks < 2; ks++) {
        FragAB a[2], b[4];
        // A fragment: lanes 0-15 hold K{0..7,16..23}, lanes 16-31 K{8..15,24..31}
        #pragma unroll
        for (int m = 0; m < 2; m++) {
          int row = rowBaseW + m * 16 + laneM;
          int kA  = ks * 32 + laneHi * 8;
          a[m].h[0] = *(const us8*)&sA[row][kA];
          a[m].h[1] = *(const us8*)&sA[row][kA + 16];
        }
        // B fragment: lanes 0-15 hold K{0..7,16..23}, lanes 16-31 K{8..15,24..31}
        #pragma unroll
        for (int n = 0; n < 4; n++) {
          int col = n * 16 + laneM;
          int kB  = ks * 32 + laneHi * 8;
          b[n].h[0] = *(const us8*)&sB[col][kB];
          b[n].h[1] = *(const us8*)&sB[col][kB + 16];
        }
        #pragma unroll
        for (int m = 0; m < 2; m++)
          #pragma unroll
          for (int n = 0; n < 4; n++)
            y[m][n] = __builtin_amdgcn_wmma_f32_16x16x32_bf16(
                false, a[m].v, false, b[n].v, (short)0, y[m][n], false, false);
      }

      // scale partial product by per-row prob * mask bit, fold into acc
      float mbit = (float)((mask >> e) & 1u);
      #pragma unroll
      for (int m = 0; m < 2; m++) {
        int rb = rowBaseW + m * 16 + laneHi * 8;
        float p[8];
        #pragma unroll
        for (int v = 0; v < 8; v++) p[v] = sP[rb + v][e] * mbit;
        #pragma unroll
        for (int n = 0; n < 4; n++)
          #pragma unroll
          for (int v = 0; v < 8; v++)
            acc[m][n][v] += p[v] * y[m][n][v];
      }

      if (e < N_EXP - 1) {
        __syncthreads();   // all waves done reading current sB
        *(us8*)&sB[bN0][bK0] = nB0;
        *(us8*)&sB[bN1][bK1] = nB1;
        __syncthreads();
      }
    }
  }

  // Store f32 output
  #pragma unroll
  for (int m = 0; m < 2; m++) {
    int rb0 = blockRow * 256 + rowBaseW + m * 16 + laneHi * 8;
    #pragma unroll
    for (int v = 0; v < 8; v++) {
      float* orow = out + (size_t)(rb0 + v) * D_MODEL + colBase + laneM;
      #pragma unroll
      for (int n = 0; n < 4; n++) orow[n * 16] = acc[m][n][v];
    }
  }
}

// ---------------------------------------------------------------------------
// Launch. Workspace layout: wbT (2MB) | probs (4MB) | xb (32MB) | mask (1KB)
// ---------------------------------------------------------------------------
extern "C" void kernel_launch(void* const* d_in, const int* in_sizes, int n_in,
                              void* d_out, int out_size, void* d_ws, size_t ws_size,
                              hipStream_t stream) {
  const float* x      = (const float*)d_in[0];
  const float* weight = (const float*)d_in[1];
  const float* gate_w = (const float*)d_in[2];
  const float* gate_b = (const float*)d_in[3];
  float* out = (float*)d_out;

  unsigned char* ws = (unsigned char*)d_ws;
  unsigned short* wbT   = (unsigned short*)(ws);                          // 2 MB
  float*          probs = (float*)(ws + (2ull << 20));                    // 4 MB
  unsigned short* xb    = (unsigned short*)(ws + (6ull << 20));           // 32 MB
  unsigned int*   bmask = (unsigned int*)(ws + (38ull << 20));            // 1 KB

  moe_prep_kernel<<<(N_EXP * D_MODEL * D_MODEL) / 256, 256, 0, stream>>>(weight, wbT, bmask);
  moe_gate_kernel<<<B_TOKENS / 16, 256, 0, stream>>>(x, gate_w, gate_b, probs, xb, bmask);
  moe_gemm_kernel<<<dim3(D_MODEL / 64, RB), 256, 0, stream>>>(xb, wbT, probs, bmask, out);
}
